// Permuter_31361851195688
// MI455X (gfx1250) — compile-verified, matrix operand order checked by reading, not run on hardware
//
#include <hip/hip_runtime.h>

// ---------------------------------------------------------------------------
// Problem constants (from reference): B=128, N=512, D=256, H=512
// ---------------------------------------------------------------------------
constexpr int Bc = 128;
constexpr int Nc = 512;
constexpr int Dc = 256;
constexpr int Hc = 512;
constexpr int N_ITERS = 20;
constexpr float TEMP = 1.0f;
constexpr float NOISE_FACTOR = 1.0f;

constexpr int ROWS = 32;            // node rows per workgroup (2 x 16 tiles)
constexpr int PANEL = 32 * 512;     // one K-panel: 32 k-rows x 512 cols (bf16)

typedef __attribute__((ext_vector_type(16))) __bf16        v16bf;
typedef __attribute__((ext_vector_type(8)))  float         v8f;
typedef __attribute__((ext_vector_type(4)))  unsigned int  v4u;

union Frag {
    v16bf v;
    v4u   u[2];
};

// f32 -> bf16 (round-to-nearest-even)
__device__ __forceinline__ unsigned short f2bf(float f) {
    union { float f; unsigned u; } c;
    c.f = f;
    unsigned r = c.u + 0x7FFFu + ((c.u >> 16) & 1u);
    return (unsigned short)(r >> 16);
}

// ---------------------------------------------------------------------------
// Pre-pass: pack W (K x 512, f32 row-major) into bf16 panels
//   Wt[p*16384 + n*32 + k] = bf16( W[(p*32+k)*512 + n] ),  p = k-panel index
// so the MLP's panel load is a straight 32KB global->LDS b128 copy.
// One-time ~2.6MB transform; cost is negligible.
// ---------------------------------------------------------------------------
__global__ __launch_bounds__(256) void pack_w_kernel(const float* __restrict__ W,
                                                     unsigned short* __restrict__ Wt,
                                                     int total) {
    int o = blockIdx.x * 256 + threadIdx.x;
    if (o >= total) return;
    int k = o & 31;
    int n = (o >> 5) & 511;
    int p = o >> 14;
    Wt[o] = f2bf(W[(size_t)(p * 32 + k) * 512 + n]);
}

// ---------------------------------------------------------------------------
// One GEMM stage: (32 x K) bf16 activations in LDS  x  (K x 512) prepacked
// bf16 weights streamed as 32KB K-panels. 8 waves; wave w owns output column
// tiles [4w, 4w+3]; each wave computes 2 row tiles x 4 col tiles.
// Fragment layout per CDNA5 wave32 spec: lanes 0-15 hold K 0-7 / 16-23,
// lanes 16-31 hold K 8-15 / 24-31 -> two 16-byte LDS loads per fragment.
// ---------------------------------------------------------------------------
__device__ __forceinline__ void gemm_stage(const unsigned short* sIn,
                                           unsigned short*       sPan,
                                           const unsigned short* __restrict__ Wt,
                                           int K, v8f acc[2][4],
                                           int tid, int lane, int wave) {
    const int row  = lane & 15;
    const int half = lane >> 4;
    for (int k0 = 0; k0 < K; k0 += 32) {
        const int p = k0 >> 5;
        // Pure b128 copy of the prepacked panel (no conversion VALU).
        {
            const v4u* __restrict__ src = (const v4u*)(Wt + (size_t)p * PANEL);
            v4u* dst = (v4u*)sPan;
            for (int i = tid; i < PANEL / 8; i += 256) dst[i] = src[i];
        }
        if (k0 + 32 < K)  // nudge next panel toward L2 (global_prefetch_b8)
            __builtin_prefetch(Wt + (size_t)(p + 1) * PANEL + tid * 8, 0, 0);
        __syncthreads();

        // A fragments for the two row tiles (shared across 4 col tiles)
        Frag a0, a1;
        {
            const v4u* pa0 = (const v4u*)(sIn + (row)      * 512 + k0 + half * 8);
            const v4u* pa1 = (const v4u*)(sIn + (16 + row) * 512 + k0 + half * 8);
            a0.u[0] = pa0[0]; a0.u[1] = pa0[2];   // +32B -> K+16 block
            a1.u[0] = pa1[0]; a1.u[1] = pa1[2];
        }

#pragma unroll
        for (int j = 0; j < 4; ++j) {
            const int c = wave * 4 + j;
            Frag bf;
            const v4u* pb = (const v4u*)(sPan + (c * 16 + row) * 32 + half * 8);
            bf.u[0] = pb[0];
            bf.u[1] = pb[2];
            acc[0][j] = __builtin_amdgcn_wmma_f32_16x16x32_bf16(
                false, a0.v, false, bf.v, (short)0, acc[0][j], false, false);
            acc[1][j] = __builtin_amdgcn_wmma_f32_16x16x32_bf16(
                false, a1.v, false, bf.v, (short)0, acc[1][j], false, false);
        }
        __syncthreads();   // protect sPan before next panel overwrite
    }
}

// relu(acc + bias) -> bf16 activation tile in LDS (32 x 512)
// C/D layout: VGPR r holds M=r (lanes 0-15) / M=r+8 (lanes 16-31), N=lane%16.
__device__ __forceinline__ void epilogue_act(const v8f acc[2][4],
                                             const float* __restrict__ bias,
                                             unsigned short* sOut,
                                             int lane, int wave) {
    const int col16 = lane & 15;
    const int half  = lane >> 4;
#pragma unroll
    for (int t = 0; t < 2; ++t) {
#pragma unroll
        for (int j = 0; j < 4; ++j) {
            const int col = (wave * 4 + j) * 16 + col16;
            const float bv = bias[col];
#pragma unroll
            for (int r = 0; r < 8; ++r) {
                const int rw = t * 16 + half * 8 + r;
                float v = acc[t][j][r] + bv;
                v = v > 0.f ? v : 0.f;
                sOut[rw * 512 + col] = f2bf(v);
            }
        }
    }
}

// ---------------------------------------------------------------------------
// Kernel 1: fused 3-layer MLP -> la = (logits + b3 + gumbel) / TEMP
// grid = (N/32, B), 256 threads (8 waves), 96KB LDS
// ---------------------------------------------------------------------------
__global__ __launch_bounds__(256) void mlp_kernel(
    const float* __restrict__ X,  const float* __restrict__ G,
    const unsigned short* __restrict__ W1t, const float* __restrict__ b1,
    const unsigned short* __restrict__ W2t, const float* __restrict__ b2,
    const unsigned short* __restrict__ W3t, const float* __restrict__ b3,
    float* __restrict__ la) {

    __shared__ unsigned short sA[ROWS * 512];   // 32 KB activation ping
    __shared__ unsigned short sH[ROWS * 512];   // 32 KB activation pong
    __shared__ unsigned short sP[PANEL];        // 32 KB weight K-panel [n][k]

    const int b    = blockIdx.y;
    const int n0   = blockIdx.x * ROWS;
    const int tid  = threadIdx.x;
    const int lane = tid & 31;
    const int wave = tid >> 5;

    // Stage input tile X[b, n0:n0+32, 0:256] as bf16 (coalesced)
    for (int i = tid; i < ROWS * 256; i += 256) {
        int r = i >> 8, c = i & 255;
        sA[r * 512 + c] = f2bf(X[((size_t)(b * Nc + n0 + r)) * Dc + c]);
    }
    // visibility guaranteed by gemm_stage's first __syncthreads()

    v8f acc[2][4];
    const v8f vz = {0.f, 0.f, 0.f, 0.f, 0.f, 0.f, 0.f, 0.f};

#pragma unroll
    for (int t = 0; t < 2; ++t)
#pragma unroll
        for (int j = 0; j < 4; ++j) acc[t][j] = vz;
    gemm_stage(sA, sP, W1t, Dc, acc, tid, lane, wave);
    epilogue_act(acc, b1, sH, lane, wave);

#pragma unroll
    for (int t = 0; t < 2; ++t)
#pragma unroll
        for (int j = 0; j < 4; ++j) acc[t][j] = vz;
    gemm_stage(sH, sP, W2t, Hc, acc, tid, lane, wave);
    epilogue_act(acc, b2, sA, lane, wave);

#pragma unroll
    for (int t = 0; t < 2; ++t)
#pragma unroll
        for (int j = 0; j < 4; ++j) acc[t][j] = vz;
    gemm_stage(sA, sP, W3t, Hc, acc, tid, lane, wave);

    // Final epilogue: la[b, n, m] = (logit + b3[m] + gumbel[b,n,m]) / TEMP
    const int col16 = lane & 15;
    const int half  = lane >> 4;
#pragma unroll
    for (int t = 0; t < 2; ++t) {
#pragma unroll
        for (int j = 0; j < 4; ++j) {
            const int m = (wave * 4 + j) * 16 + col16;
            const float b3v = b3[m];
#pragma unroll
            for (int r = 0; r < 8; ++r) {
                const int n = n0 + t * 16 + half * 8 + r;
                const size_t idx = ((size_t)(b * Nc + n)) * Nc + m;
                la[idx] = (acc[t][j][r] + b3v + NOISE_FACTOR * G[idx]) * (1.0f / TEMP);
            }
        }
    }
}

// ---------------------------------------------------------------------------
// Kernel 2: potential-form Gumbel-Sinkhorn.
// Invariant: la_k = la0 - R[r] - C[c]  (row/col subtractions preserve this),
// so each iteration is exactly:
//   R[r] = LSE_c(la0[r][c] - C[c]);  C[c] = LSE_r(la0[r][c] - R[r])
// -> la0 is READ-ONLY (L2-resident), only 4KB of potentials live in LDS.
// grid = B, 512 threads (16 waves). Output fuses exp + (0,2,1) transpose.
// ---------------------------------------------------------------------------
__global__ __launch_bounds__(512) void sinkhorn_kernel(const float* __restrict__ la,
                                                       float* __restrict__ out) {
    __shared__ float sR[Nc];
    __shared__ float sC[Nc];

    const int b = blockIdx.x;
    const float* A = la + (size_t)b * Nc * Nc;
    float* O = out + (size_t)b * Nc * Nc;
    const int tid  = threadIdx.x;
    const int lane = tid & 31;
    const int wave = tid >> 5;
    const float NEG = -__builtin_huge_valf();

    sC[tid] = 0.f;
    __syncthreads();

    for (int it = 0; it < N_ITERS; ++it) {
        // ---- row potentials: one wave per row, coalesced, online LSE ----
#pragma unroll 1
        for (int rr = 0; rr < 32; ++rr) {
            const int r = wave * 32 + rr;
            const float* R = A + (size_t)r * Nc;
            float m = NEG, s = 0.f;
#pragma unroll
            for (int j = 0; j < 16; ++j) {
                const int c = lane + 32 * j;
                const float x = R[c] - sC[c];
                const float nm = fmaxf(m, x);
                s = s * __expf(m - nm) + __expf(x - nm);
                m = nm;
            }
#pragma unroll
            for (int off = 16; off > 0; off >>= 1) {
                const float om = __shfl_xor(m, off, 32);
                const float os = __shfl_xor(s, off, 32);
                const float nm = fmaxf(m, om);
                s = s * __expf(m - nm) + os * __expf(om - nm);
                m = nm;
            }
            if (lane == 0) sR[r] = m + __logf(s);
        }
        __syncthreads();

        // ---- col potentials: one thread per column, fully coalesced ----
        {
            float m = NEG, s = 0.f;
            for (int r = 0; r < Nc; ++r) {
                const float x = A[(size_t)r * Nc + tid] - sR[r];
                const float nm = fmaxf(m, x);
                s = s * __expf(m - nm) + __expf(x - nm);
                m = nm;
            }
            sC[tid] = m + __logf(s);
        }
        __syncthreads();
    }

    // out[b, m, n] = exp(la0[n][m] - R[n] - C[m]) — coalesced writes over n=tid
    const float Rn = sR[tid];
    for (int mm = 0; mm < Nc; ++mm)
        O[(size_t)mm * Nc + tid] = __expf(A[(size_t)tid * Nc + mm] - Rn - sC[mm]);
}

// ---------------------------------------------------------------------------
extern "C" void kernel_launch(void* const* d_in, const int* in_sizes, int n_in,
                              void* d_out, int out_size, void* d_ws, size_t ws_size,
                              hipStream_t stream) {
    (void)in_sizes; (void)n_in; (void)out_size; (void)ws_size;
    const float* X  = (const float*)d_in[0];
    const float* G  = (const float*)d_in[1];
    const float* W1 = (const float*)d_in[2];
    const float* b1 = (const float*)d_in[3];
    const float* W2 = (const float*)d_in[4];
    const float* b2 = (const float*)d_in[5];
    const float* W3 = (const float*)d_in[6];
    const float* b3 = (const float*)d_in[7];
    float* out = (float*)d_out;

    // Workspace layout: [ la : B*N*N f32 | W1t | W2t | W3t (bf16 panels) ]
    char* ws = (char*)d_ws;
    float* la = (float*)ws;
    size_t laBytes = (size_t)Bc * Nc * Nc * sizeof(float);       // 128 MiB
    unsigned short* W1t = (unsigned short*)(ws + laBytes);
    unsigned short* W2t = W1t + (size_t)Dc * Hc;                 // 256*512
    unsigned short* W3t = W2t + (size_t)Hc * Hc;                 // 512*512

    // One-time weight packing (f32 -> bf16, panel-major)
    pack_w_kernel<<<(Dc * Hc) / 256, 256, 0, stream>>>(W1, W1t, Dc * Hc);
    pack_w_kernel<<<(Hc * Hc) / 256, 256, 0, stream>>>(W2, W2t, Hc * Hc);
    pack_w_kernel<<<(Hc * Nc) / 256, 256, 0, stream>>>(W3, W3t, Hc * Nc);

    dim3 g1(Nc / ROWS, Bc);
    mlp_kernel<<<g1, 256, 0, stream>>>(X, G, W1t, b1, W2t, b2, W3t, b3, la);
    sinkhorn_kernel<<<Bc, 512, 0, stream>>>(la, out);
}
